// Decoder_80221399155193
// MI455X (gfx1250) — compile-verified, matrix operand order checked by reading, not run on hardware
//
#include <hip/hip_runtime.h>
#include <hip/hip_bf16.h>

// ---------------- problem constants ----------------
#define Bb     128
#define Tt     512
#define KVd    128
#define EMBd   256
#define H1d    512
#define VOCABd 30
#define Ld     256
#define K1d    (EMBd + KVd + H1d)   // 896  (x = [emb | ctx | h1])
#define N1d    (4 * H1d)            // 2048 gates1
#define K2d    (H1d + KVd)          // 640  (a2 = [h1 | h2])
#define N2d    (4 * KVd)            // 512  gates2
#define NWG    16
#define NTHR   256
#define PREDSZ (Bb * Ld * VOCABd)

// LDS-resident W1 slice: 128 rows x 896, padded to 920 bf16/row so the
// 16-row fragment access pattern is bank-conflict-free (460 dw, 460%64=12).
#define LW1P   920
#define LDS_W1_BYTES   (128 * LW1P * 2)                 // 235,520
#define LDS_OFF_H2     LDS_W1_BYTES                     // 8*128 f32
#define LDS_OFF_ATTN   (LDS_OFF_H2 + 8 * KVd * 4)       // 8*512 f32
#define LDS_OFF_CTX    (LDS_OFF_ATTN + 8 * Tt * 4)      // 8*128 f32
#define SMEM_BYTES     (LDS_OFF_CTX + 8 * KVd * 4)      // 260,096 total

typedef __bf16 bf16_t;
typedef __attribute__((ext_vector_type(16))) __bf16 v16bf;
typedef __attribute__((ext_vector_type(8)))  float  v8f;

// ---------------- workspace layout (bytes) ----------------
static constexpr size_t OFF_W1   = 0;                                   // 2048*896 bf16
static constexpr size_t OFF_W2   = OFF_W1  + (size_t)N1d * K1d * 2;     // 512*640 bf16
static constexpr size_t OFF_XCAT = OFF_W2  + (size_t)N2d * K2d * 2;     // 128*896 bf16
static constexpr size_t OFF_A2   = OFF_XCAT+ (size_t)Bb  * K1d * 2;     // 128*640 bf16
static constexpr size_t OFF_C1   = OFF_A2  + (size_t)Bb  * K2d * 2;     // 128*512 f32
static constexpr size_t OFF_C2   = OFF_C1  + (size_t)Bb  * H1d * 4;     // 128*128 f32
static constexpr size_t OFF_H2   = OFF_C2  + (size_t)Bb  * KVd * 4;     // 128*128 f32
static constexpr size_t OFF_B1   = OFF_H2  + (size_t)Bb  * KVd * 4;     // 2048 f32
static constexpr size_t OFF_B2   = OFF_B1  + (size_t)N1d * 4;           // 512 f32
static constexpr size_t OFF_BAR  = OFF_B2  + (size_t)N2d * 4;           // barrier u32

__device__ __forceinline__ float sigm(float x) { return 1.0f / (1.0f + __expf(-x)); }

// Assemble a 32B WMMA fragment from a 16B-aligned source (LDS or global)
// using two explicit b128 loads; avoids over-aligned vector deref.
struct frag_bits { uint4 a, b; };
__device__ __forceinline__ v16bf load_frag16(const bf16_t* p) {
  frag_bits t;
  t.a = *(const uint4*)p;
  t.b = *(const uint4*)(p + 8);
  return __builtin_bit_cast(v16bf, t);
}

// ---------------- init: bf16 weight repack, fused biases, zero states ----------------
__global__ void decoder_init_kernel(
    const float* __restrict__ Wih1, const float* __restrict__ Whh1,
    const float* __restrict__ bih1, const float* __restrict__ bhh1,
    const float* __restrict__ Wih2, const float* __restrict__ Whh2,
    const float* __restrict__ bih2, const float* __restrict__ bhh2,
    unsigned short* ws_w1, unsigned short* ws_w2,
    unsigned short* ws_xcat, unsigned short* ws_a2,
    float* c1, float* c2, float* h2f, float* bias1, float* bias2, unsigned* bar)
{
  bf16_t* w1   = (bf16_t*)ws_w1;
  bf16_t* w2   = (bf16_t*)ws_w2;
  bf16_t* xcat = (bf16_t*)ws_xcat;
  bf16_t* a2   = (bf16_t*)ws_a2;
  const int idx    = blockIdx.x * blockDim.x + threadIdx.x;
  const int stride = gridDim.x * blockDim.x;

  // W1cat (N1 x K1): [W_ih1 | W_hh1], row-major == WMMA B-operand layout
  #pragma unroll 1
  for (int i = idx; i < N1d * K1d; i += stride) {
    int n = i / K1d, k = i % K1d;
    float v = (k < EMBd + KVd) ? Wih1[(size_t)n * (EMBd + KVd) + k]
                               : Whh1[(size_t)n * H1d + (k - (EMBd + KVd))];
    w1[i] = (bf16_t)v;
  }
  // W2cat (N2 x K2): [W_ih2 | W_hh2]
  #pragma unroll 1
  for (int i = idx; i < N2d * K2d; i += stride) {
    int n = i / K2d, k = i % K2d;
    float v = (k < H1d) ? Wih2[(size_t)n * H1d + k]
                        : Whh2[(size_t)n * KVd + (k - H1d)];
    w2[i] = (bf16_t)v;
  }
  for (int i = idx; i < N1d; i += stride) bias1[i] = bih1[i] + bhh1[i];
  for (int i = idx; i < N2d; i += stride) bias2[i] = bih2[i] + bhh2[i];
  #pragma unroll 1
  for (int i = idx; i < Bb * K1d; i += stride) xcat[i] = (bf16_t)0.0f;
  #pragma unroll 1
  for (int i = idx; i < Bb * K2d; i += stride) a2[i]   = (bf16_t)0.0f;
  #pragma unroll 1
  for (int i = idx; i < Bb * H1d; i += stride) c1[i]   = 0.0f;
  #pragma unroll 1
  for (int i = idx; i < Bb * KVd; i += stride) { c2[i] = 0.0f; h2f[i] = 0.0f; }
  if (idx == 0) bar[0] = 0u;
}

// ---------------- device-scope barrier for the persistent kernel ----------------
__device__ __forceinline__ void global_barrier(unsigned* bar, unsigned& target)
{
  __threadfence();          // release: make this WG's global writes visible
  __syncthreads();
  target += NWG;
  if (threadIdx.x == 0) {
    __hip_atomic_fetch_add(bar, 1u, __ATOMIC_RELEASE, __HIP_MEMORY_SCOPE_AGENT);
    while (__hip_atomic_load(bar, __ATOMIC_ACQUIRE, __HIP_MEMORY_SCOPE_AGENT) < target) {
      __builtin_amdgcn_s_sleep(2);
    }
  }
  __syncthreads();
  __threadfence();          // acquire: drop stale lines before reading peers' data
}

// ---------------- persistent decoder: all 256 steps in one kernel ----------------
__global__ __launch_bounds__(NTHR, 1) void decoder_kernel(
    const float* __restrict__ key, const float* __restrict__ value,
    const int*   __restrict__ enc_len, const int* __restrict__ y,
    const float* __restrict__ embW, const float* __restrict__ out_bias,
    unsigned short* __restrict__ ws_w1, unsigned short* __restrict__ ws_w2,
    unsigned short* __restrict__ ws_xcat, unsigned short* __restrict__ ws_a2,
    float* __restrict__ c1, float* __restrict__ c2, float* __restrict__ h2f,
    const float* __restrict__ bias1, const float* __restrict__ bias2,
    unsigned* __restrict__ bar, float* __restrict__ out)
{
  extern __shared__ __align__(16) char dyn_smem[];
  bf16_t* lw1     = (bf16_t*)dyn_smem;                    // WG's W1 slice, all 256 steps
  float*  lds_h2  = (float*)(dyn_smem + LDS_OFF_H2);      // [8][128]
  float*  lds_attn= (float*)(dyn_smem + LDS_OFF_ATTN);    // [8][512]
  float*  lds_ctx = (float*)(dyn_smem + LDS_OFF_CTX);     // [8][128]

  const int g    = blockIdx.x;        // 0..15
  const int tid  = threadIdx.x;
  const int wid  = tid >> 5;          // 0..7
  const int lane = tid & 31;
  const int lr   = lane & 15;         // fragment row/col within tile
  const int lh   = lane >> 4;         // K-half select
  const int kofl = lh * 16;           // per-lane K offset inside 32-wide K tile

  const bf16_t* w1 = (const bf16_t*)ws_w1;
  const bf16_t* w2 = (const bf16_t*)ws_w2;
  bf16_t* xcat = (bf16_t*)ws_xcat;    // A1 = [emb | ctx | h1] (B x 896), bf16
  bf16_t* a2   = (bf16_t*)ws_a2;      // A2 = [h1 | h2]        (B x 640), bf16

  // ---- one-time: stage this WG's 128 W1 rows (4 gates x 32 cols) into LDS ----
  // local row rr = q*32 + j  <->  global row q*512 + g*32 + j
  #pragma unroll 1
  for (int rr = wid; rr < 128; rr += 8) {
    const int q = rr >> 5, j = rr & 31;
    const bf16_t* srow = w1 + (size_t)(q * H1d + g * 32 + j) * K1d;
    bf16_t*       drow = lw1 + (size_t)rr * LW1P;
    #pragma unroll 1
    for (int k = lane * 8; k < K1d; k += 32 * 8)
      *(uint4*)(drow + k) = *(const uint4*)(srow + k);
  }
  __syncthreads();

  unsigned target = 0;

  #pragma unroll 1
  for (int step = 0; step < Ld; ++step) {
    // ---- phase 0: teacher-forcing embedding into xcat[:,0:256] (dup per WG, benign) ----
    #pragma unroll 1
    for (int i = tid; i < Bb * EMBd; i += NTHR) {
      int bb = i >> 8, ee = i & (EMBd - 1);
      float v = 0.0f;
      if (step > 0) { int tok = y[bb * Ld + step - 1]; v = embW[(size_t)tok * EMBd + ee]; }
      xcat[(size_t)bb * K1d + ee] = (bf16_t)v;
    }
    __threadfence_block();
    __syncthreads();

    // ---- phase 1: GEMM1 (128x896 @ 896x2048) + LSTM1 cell fused in registers ----
    // B-fragments come from LDS (conflict-free); A-fragments from L2-hot xcat.
    {
      const int m0 = wid * 16;
      v8f acc[4][2];
      #pragma unroll
      for (int q = 0; q < 4; ++q)
        #pragma unroll
        for (int s = 0; s < 2; ++s) acc[q][s] = v8f{};

      const bf16_t* arow = xcat + (size_t)(m0 + lr) * K1d + kofl;
      const bf16_t* brow = lw1 + (size_t)lr * LW1P + kofl;
      #pragma unroll 1                       // keep live set small: no k-unroll
      for (int k0 = 0; k0 < K1d; k0 += 32) {
        v16bf afrag = load_frag16(arow + k0);
        #pragma unroll
        for (int q = 0; q < 4; ++q) {
          #pragma unroll
          for (int s = 0; s < 2; ++s) {
            v16bf bfrag = load_frag16(brow + (size_t)(q * 32 + s * 16) * LW1P + k0);
            acc[q][s] = __builtin_amdgcn_wmma_f32_16x16x32_bf16(
                false, afrag, false, bfrag, (short)0, acc[q][s], false, false);
          }
        }
      }
      #pragma unroll
      for (int s = 0; s < 2; ++s) {
        const int jj = g * 32 + s * 16 + lr;          // h1 column
        const float bi = bias1[jj];
        const float bf_ = bias1[H1d + jj];
        const float bg = bias1[2 * H1d + jj];
        const float bo = bias1[3 * H1d + jj];
        #pragma unroll
        for (int r = 0; r < 8; ++r) {
          const int m = m0 + lh * 8 + r;              // batch row
          float iv = acc[0][s][r] + bi;
          float fv = acc[1][s][r] + bf_;
          float gv = acc[2][s][r] + bg;
          float ov = acc[3][s][r] + bo;
          float cp = c1[(size_t)m * H1d + jj];
          float cn = sigm(fv) * cp + sigm(iv) * tanhf(gv);
          float hn = sigm(ov) * tanhf(cn);
          c1[(size_t)m * H1d + jj] = cn;
          bf16_t hb = (bf16_t)hn;
          xcat[(size_t)m * K1d + (EMBd + KVd) + jj] = hb;   // feeds next step's GEMM1
          a2[(size_t)m * K2d + jj] = hb;                    // feeds GEMM2 this step
        }
      }
    }
    global_barrier(bar, target);   // all of h1 visible

    // ---- phase 2: GEMM2 (128x640 @ 640x512) + LSTM2 cell, WGs 0..7 only ----
    if (g < 8) {
      const int m0 = wid * 16;
      v8f acc2[4];
      #pragma unroll
      for (int q = 0; q < 4; ++q) acc2[q] = v8f{};

      const bf16_t* arow = a2 + (size_t)(m0 + lr) * K2d + kofl;
      const bf16_t* brow = w2 + (size_t)(g * 16 + lr) * K2d + kofl;
      #pragma unroll 1
      for (int k0 = 0; k0 < K2d; k0 += 32) {
        v16bf afrag = load_frag16(arow + k0);
        #pragma unroll
        for (int q = 0; q < 4; ++q) {
          v16bf bfrag = load_frag16(brow + (size_t)(q * KVd) * K2d + k0);
          acc2[q] = __builtin_amdgcn_wmma_f32_16x16x32_bf16(
              false, afrag, false, bfrag, (short)0, acc2[q], false, false);
        }
      }
      const int jj = g * 16 + lr;                     // h2 column
      const float bi = bias2[jj];
      const float bf_ = bias2[KVd + jj];
      const float bg = bias2[2 * KVd + jj];
      const float bo = bias2[3 * KVd + jj];
      #pragma unroll
      for (int r = 0; r < 8; ++r) {
        const int m = m0 + lh * 8 + r;
        float iv = acc2[0][r] + bi;
        float fv = acc2[1][r] + bf_;
        float gv = acc2[2][r] + bg;
        float ov = acc2[3][r] + bo;
        float cp = c2[(size_t)m * KVd + jj];
        float cn = sigm(fv) * cp + sigm(iv) * tanhf(gv);
        float hn = sigm(ov) * tanhf(cn);
        c2[(size_t)m * KVd + jj] = cn;
        h2f[(size_t)m * KVd + jj] = hn;               // f32 query for attention/pred
        a2[(size_t)m * K2d + H1d + jj] = (bf16_t)hn;  // feeds next step's GEMM2
      }
    }
    global_barrier(bar, target);   // h2 visible

    // ---- phase 3: attention + output projection, one wave per batch row ----
    // Energies live in LDS; softmax is three unroll-1 sweeps + shuffle reductions.
    {
      const int b = g * 8 + wid;                      // 0..127
      for (int d = lane; d < KVd; d += 32) lds_h2[wid * KVd + d] = h2f[(size_t)b * KVd + d];
      __syncthreads();

      const int elen = enc_len[b];
      const float* kb = key + (size_t)b * Tt * KVd;
      float mx = -3.4e38f;
      #pragma unroll 1
      for (int c = 0; c < 16; ++c) {
        const int ttp = c * 32 + lane;
        const float* kr = kb + (size_t)ttp * KVd;
        __builtin_prefetch(kr + 32 * KVd, 0, 1);      // next t-chunk (global_prefetch)
        float s = 0.0f;
        #pragma unroll 8
        for (int d = 0; d < KVd; d += 4) {
          float4 k4 = *(const float4*)(kr + d);
          s += k4.x * lds_h2[wid * KVd + d]     + k4.y * lds_h2[wid * KVd + d + 1]
             + k4.z * lds_h2[wid * KVd + d + 2] + k4.w * lds_h2[wid * KVd + d + 3];
        }
        s = (ttp >= elen) ? -1.0e9f : s;
        s *= 0.08838834764831845f;                    // 1/sqrt(128)
        lds_attn[wid * Tt + ttp] = s;
        mx = fmaxf(mx, s);
      }
      #pragma unroll
      for (int off = 16; off > 0; off >>= 1) mx = fmaxf(mx, __shfl_xor(mx, off));
      float sum = 0.0f;
      #pragma unroll 1
      for (int c = 0; c < 16; ++c) {
        const int ttp = c * 32 + lane;
        float vexp = __expf(lds_attn[wid * Tt + ttp] - mx);
        sum += vexp;
        lds_attn[wid * Tt + ttp] = vexp;
      }
      #pragma unroll
      for (int off = 16; off > 0; off >>= 1) sum += __shfl_xor(sum, off);
      const float inv = 1.0f / sum;
      #pragma unroll 1
      for (int c = 0; c < 16; ++c) lds_attn[wid * Tt + c * 32 + lane] *= inv;
      __syncthreads();

      // ctx = attn @ value ; lane owns 4 consecutive d -> fully coalesced value rows
      const float* vb = value + (size_t)b * Tt * KVd;
      float cx0 = 0.f, cx1 = 0.f, cx2 = 0.f, cx3 = 0.f;
      #pragma unroll 2
      for (int ttp = 0; ttp < Tt; ++ttp) {
        const float a = lds_attn[wid * Tt + ttp];     // LDS broadcast
        float4 v4 = *(const float4*)(vb + (size_t)ttp * KVd + lane * 4);
        cx0 += a * v4.x; cx1 += a * v4.y; cx2 += a * v4.z; cx3 += a * v4.w;
      }
      {
        const int d0 = lane * 4;
        lds_ctx[wid * KVd + d0]     = cx0; lds_ctx[wid * KVd + d0 + 1] = cx1;
        lds_ctx[wid * KVd + d0 + 2] = cx2; lds_ctx[wid * KVd + d0 + 3] = cx3;
        xcat[(size_t)b * K1d + EMBd + d0]     = (bf16_t)cx0;
        xcat[(size_t)b * K1d + EMBd + d0 + 1] = (bf16_t)cx1;
        xcat[(size_t)b * K1d + EMBd + d0 + 2] = (bf16_t)cx2;
        xcat[(size_t)b * K1d + EMBd + d0 + 3] = (bf16_t)cx3;
      }
      if (b == 0) {
        #pragma unroll 1
        for (int ttp = lane; ttp < Tt; ttp += 32)
          out[PREDSZ + (size_t)step * Tt + ttp] = lds_attn[wid * Tt + ttp];
      }
      __syncthreads();

      // pred[b,v] = [h2|ctx] . embW[v,:] + out_b[v]   (lane == vocab index)
      if (lane < VOCABd) {
        float s = out_bias[lane];
        #pragma unroll 4
        for (int ee = 0; ee < KVd; ++ee) s += lds_h2[wid * KVd + ee]  * embW[(size_t)lane * EMBd + ee];
        #pragma unroll 4
        for (int ee = 0; ee < KVd; ++ee) s += lds_ctx[wid * KVd + ee] * embW[(size_t)lane * EMBd + KVd + ee];
        out[((size_t)b * Ld + step) * VOCABd + lane] = s;
      }
    }
    global_barrier(bar, target);   // ctx visible for next step's GEMM1
  }
}

// ---------------- host entry ----------------
extern "C" void kernel_launch(void* const* d_in, const int* in_sizes, int n_in,
                              void* d_out, int out_size, void* d_ws, size_t ws_size,
                              hipStream_t stream) {
  (void)in_sizes; (void)n_in; (void)out_size; (void)ws_size;
  const float* key     = (const float*)d_in[0];
  const float* value   = (const float*)d_in[1];
  const int*   enc_len = (const int*)d_in[2];
  const int*   y       = (const int*)d_in[3];
  const float* embW    = (const float*)d_in[4];
  const float* Wih1    = (const float*)d_in[5];
  const float* Whh1    = (const float*)d_in[6];
  const float* bih1    = (const float*)d_in[7];
  const float* bhh1    = (const float*)d_in[8];
  const float* Wih2    = (const float*)d_in[9];
  const float* Whh2    = (const float*)d_in[10];
  const float* bih2    = (const float*)d_in[11];
  const float* bhh2    = (const float*)d_in[12];
  const float* out_b   = (const float*)d_in[13];

  char* ws = (char*)d_ws;
  unsigned short* w1_p   = (unsigned short*)(ws + OFF_W1);
  unsigned short* w2_p   = (unsigned short*)(ws + OFF_W2);
  unsigned short* xcat_p = (unsigned short*)(ws + OFF_XCAT);
  unsigned short* a2_p   = (unsigned short*)(ws + OFF_A2);
  float* c1_p   = (float*)(ws + OFF_C1);
  float* c2_p   = (float*)(ws + OFF_C2);
  float* h2_p   = (float*)(ws + OFF_H2);
  float* b1_p   = (float*)(ws + OFF_B1);
  float* b2_p   = (float*)(ws + OFF_B2);
  unsigned* bar = (unsigned*)(ws + OFF_BAR);

  // Allow >64KB dynamic LDS (CDNA5: up to 320KB per workgroup).
  (void)hipFuncSetAttribute((const void*)decoder_kernel,
                            hipFuncAttributeMaxDynamicSharedMemorySize,
                            (int)SMEM_BYTES);

  decoder_init_kernel<<<1024, 256, 0, stream>>>(
      Wih1, Whh1, bih1, bhh1, Wih2, Whh2, bih2, bhh2,
      w1_p, w2_p, xcat_p, a2_p, c1_p, c2_p, h2_p, b1_p, b2_p, bar);

  decoder_kernel<<<NWG, NTHR, SMEM_BYTES, stream>>>(
      key, value, enc_len, y, embW, out_b,
      w1_p, w2_p, xcat_p, a2_p, c1_p, c2_p, h2_p, b1_p, b2_p, bar,
      (float*)d_out);
}
